// RRTAttention_48661979464003
// MI455X (gfx1250) — matrix operand order repaired and checked by reading, not compiled, not run on hardware
//
#include <hip/hip_runtime.h>
#include <hip/hip_bf16.h>
#include <math.h>

// ---------------- problem constants ----------------
#define B_   2
#define T_   2048
#define D_   1024
#define NH_  16
#define NKV_ 4
#define HD_  64
#define KD_  (NKV_*HD_)       // 256
#define PD_  16
#define RANK_ 8
#define BT_  (B_*T_)          // 4096
#define NQKV_ (D_ + 2*KD_)    // 1536

#define NEG_INF (-__builtin_inff())

typedef __attribute__((ext_vector_type(16))) __bf16 v16bf;
typedef __attribute__((ext_vector_type(8)))  float  v8f;

union FragBF { v16bf v; unsigned int u[8]; };

// fp32 -> bf16 (round to nearest even)
__device__ __forceinline__ unsigned short f2bf(float f) {
  union { float f; unsigned u; } c; c.f = f;
  unsigned r = c.u + 0x7FFFu + ((c.u >> 16) & 1u);
  return (unsigned short)(r >> 16);
}

// A-fragment: bf16 16(M)x32(K). Lanes 0-15 rows M=lane: VGPR p<4 -> K=2p,
// VGPR p>=4 -> K=16+2(p-4); lanes 16-31 same rows with K offset +8.
__device__ __forceinline__ void load_a_frag(FragBF& f, const unsigned short* A,
                                            int lda, int m0, int k0, int lane) {
  int row = lane & 15, lh = lane >> 4;
  const unsigned int* p32 =
      (const unsigned int*)(A + (size_t)(m0 + row) * (size_t)lda + k0 + 8 * lh);
  f.u[0] = p32[0]; f.u[1] = p32[1]; f.u[2] = p32[2]; f.u[3] = p32[3];
  f.u[4] = p32[8]; f.u[5] = p32[9]; f.u[6] = p32[10]; f.u[7] = p32[11];
}

// B-fragment: bf16 32(K)x16(N), Bt stored as [N][ldb] row-major (i.e. B^T).
// Lanes 0-15: N=lane, VGPR p<4 -> K=2p, VGPR p>=4 -> K=16+2(p-4); lanes 16-31: K offset +8.
__device__ __forceinline__ void load_b_frag(FragBF& f, const unsigned short* Bt,
                                            size_t ldb, int n0, int k0, int lane) {
  int col = lane & 15, lh = lane >> 4;
  const unsigned int* p32 =
      (const unsigned int*)(Bt + (size_t)(n0 + col) * ldb + k0 + 8 * lh);
  f.u[0] = p32[0]; f.u[1] = p32[1]; f.u[2] = p32[2]; f.u[3] = p32[3];
  f.u[4] = p32[8]; f.u[5] = p32[9]; f.u[6] = p32[10]; f.u[7] = p32[11];
}

__device__ __forceinline__ v8f wmma_bf16(const FragBF& a, const FragBF& b, v8f c) {
  return __builtin_amdgcn_wmma_f32_16x16x32_bf16(false, a.v, false, b.v,
                                                 (short)0, c, false, false);
}

// ---------------- kernel: fp32 -> bf16 convert ----------------
__global__ void f32_to_bf16_kernel(const float* __restrict__ src,
                                   unsigned short* __restrict__ dst, size_t n) {
  size_t i = (size_t)blockIdx.x * blockDim.x + threadIdx.x;
  if (i < n) dst[i] = f2bf(src[i]);
}

// ---------------- kernel: fold LoRA into concatenated QKV weight ----------------
__global__ void build_wall_kernel(const float* __restrict__ Wq,
                                  const float* __restrict__ Wk,
                                  const float* __restrict__ Wv,
                                  const float* __restrict__ Aq,
                                  const float* __restrict__ Bq,
                                  const float* __restrict__ Av,
                                  const float* __restrict__ Bv,
                                  unsigned short* __restrict__ Wall) {
  int idx = blockIdx.x * blockDim.x + threadIdx.x;
  if (idx >= NQKV_ * D_) return;
  int o = idx / D_, i = idx % D_;
  float val;
  if (o < D_) {
    val = Wq[(size_t)o * D_ + i];
#pragma unroll
    for (int r = 0; r < RANK_; ++r)
      val += Bq[(size_t)o * RANK_ + r] * Aq[(size_t)r * D_ + i];
  } else if (o < D_ + KD_) {
    val = Wk[(size_t)(o - D_) * D_ + i];
  } else {
    int oo = o - D_ - KD_;
    val = Wv[(size_t)oo * D_ + i];
#pragma unroll
    for (int r = 0; r < RANK_; ++r)
      val += Bv[(size_t)oo * RANK_ + r] * Av[(size_t)r * D_ + i];
  }
  Wall[idx] = f2bf(val);
}

// ---------------- kernel: WMMA GEMM  C[M,N] = A[M,K] * Bt[N,K]^T ----------------
// bf16 in / fp32 out. 8 waves/block; wave tile 32(M)x64(N); block tile 64x256.
// Register double-buffered k-loop (requires K % 64 == 0). launch_bounds(256,1)
// raises the VGPR budget so acc(64) + 2x6 fragments(96) stay in registers.
__global__ void __launch_bounds__(256, 1)
gemm_bf16_f32(const unsigned short* __restrict__ A,
              const unsigned short* __restrict__ Bt,
              float* __restrict__ C, int M, int N, int K) {
  int lane = threadIdx.x & 31, wave = threadIdx.x >> 5;
  int m0 = blockIdx.x * 64 + (wave & 1) * 32;
  int n0 = blockIdx.y * 256 + (wave >> 1) * 64;
  if (m0 >= M || n0 >= N) return;

  v8f acc[2][4] = {};
  FragBF a0[2], b0[4], a1[2], b1[4];

  auto load_tile = [&](FragBF (&a)[2], FragBF (&b)[4], int k0) {
#pragma unroll
    for (int i = 0; i < 2; ++i) load_a_frag(a[i], A, K, m0 + 16 * i, k0, lane);
#pragma unroll
    for (int j = 0; j < 4; ++j) load_b_frag(b[j], Bt, (size_t)K, n0 + 16 * j, k0, lane);
  };
  auto compute = [&](FragBF (&a)[2], FragBF (&b)[4]) {
#pragma unroll
    for (int i = 0; i < 2; ++i)
#pragma unroll
      for (int j = 0; j < 4; ++j)
        acc[i][j] = wmma_bf16(a[i], b[j], acc[i][j]);
  };

  int nk = K >> 5;                       // k-steps of 32 (nk even, >= 2)
  load_tile(a0, b0, 0);
  int i = 0;
  for (; i < nk - 2; i += 2) {
    int kpf = (i << 5) + 128;            // prefetch 4 steps ahead (near scope)
    if (kpf < K) {
      __builtin_prefetch(A + (size_t)(m0 + lane) * K + kpf, 0, 3);
      __builtin_prefetch(Bt + (size_t)(n0 + lane) * K + kpf, 0, 3);
      __builtin_prefetch(Bt + (size_t)(n0 + 32 + lane) * K + kpf, 0, 3);
    }
    load_tile(a1, b1, (i + 1) << 5);
    compute(a0, b0);
    load_tile(a0, b0, (i + 2) << 5);
    compute(a1, b1);
  }
  load_tile(a1, b1, (nk - 1) << 5);      // peeled final stage
  compute(a0, b0);
  compute(a1, b1);

  int col = lane & 15, lh = lane >> 4;
#pragma unroll
  for (int i2 = 0; i2 < 2; ++i2)
#pragma unroll
    for (int j = 0; j < 4; ++j)
#pragma unroll
      for (int v = 0; v < 8; ++v)
        C[(size_t)(m0 + 16 * i2 + v + 8 * lh) * (size_t)N + n0 + 16 * j + col] =
            acc[i2][j][v];
}

// ---------------- kernel: RMS-norm + partial RoPE + gain + layout ----------------
__global__ void qkv_postprocess(const float* __restrict__ qkv,
                                const float* __restrict__ q_gain,
                                unsigned short* __restrict__ Qb,
                                unsigned short* __restrict__ Kb,
                                unsigned short* __restrict__ VTb) {
  int tok = blockIdx.x;                 // 0..BT-1
  int b = tok / T_, t = tok % T_;
  int lane = threadIdx.x & 31, wave = threadIdx.x >> 5;
  const float* row = qkv + (size_t)tok * NQKV_;
#pragma unroll
  for (int it = 0; it < 3; ++it) {
    int hidx = wave + it * 8;           // 0..23 : 16 q heads, 4 k heads, 4 v heads
    int kind, h, base;
    if (hidx < NH_)             { kind = 0; h = hidx;              base = h * HD_; }
    else if (hidx < NH_ + NKV_) { kind = 1; h = hidx - NH_;        base = D_ + h * HD_; }
    else                        { kind = 2; h = hidx - NH_ - NKV_; base = D_ + KD_ + h * HD_; }
    float a  = row[base + lane];
    float c2 = row[base + lane + 32];
    if (kind < 2) {
      float ss = a * a + c2 * c2;
#pragma unroll
      for (int off = 16; off >= 1; off >>= 1) ss += __shfl_xor(ss, off, 32);
      float r = rsqrtf(ss * (1.0f / HD_) + 1.1920929e-7f);
      a *= r; c2 *= r;
      float other = __shfl_xor(a, 8, 32);
      if (lane < 16) {
        int j = lane & 7;
        float fr = (float)t * powf(10000.0f, -(float)j * 0.125f);
        float cs = cosf(fr), sn = sinf(fr);
        a = (lane < 8) ? (a * cs + other * sn) : (-a * cs + other * sn);
      }
      if (kind == 0) {
        float g = q_gain[h];
        a *= g; c2 *= g;
        unsigned short* dst = Qb + ((size_t)(b * NH_ + h) * T_ + t) * HD_;
        dst[lane] = f2bf(a); dst[lane + 32] = f2bf(c2);
      } else {
        unsigned short* dst = Kb + ((size_t)(b * NKV_ + h) * T_ + t) * HD_;
        dst[lane] = f2bf(a); dst[lane + 32] = f2bf(c2);
      }
    } else {
      unsigned short* dst = VTb + ((size_t)(b * NKV_ + h) * HD_) * T_;
      dst[(size_t)lane * T_ + t]        = f2bf(a);
      dst[(size_t)(lane + 32) * T_ + t] = f2bf(c2);
    }
  }
}

// ---------------- kernel: causal flash attention (WMMA) ----------------
// One wave per 16-row Q tile; key tiles of 32 so P is exactly one 16x32 A-frag.
// K-fragments double-buffered across key tiles; V-fragments issued before the
// softmax VALU block so shuffle/exp work hides their latency.
__global__ void __launch_bounds__(256, 1)
flash_attn(const unsigned short* __restrict__ Qb,
           const unsigned short* __restrict__ Kb,
           const unsigned short* __restrict__ VTb,
           unsigned short* __restrict__ Y) {
  __shared__ unsigned short Ptile[8][16 * 32];   // per-wave P staging, 8 KB
  int lane = threadIdx.x & 31, wave = threadIdx.x >> 5;
  const int tiles128 = T_ / 128;
  int bh = blockIdx.x / tiles128;
  int t128 = blockIdx.x % tiles128;
  int b = bh / NH_, h = bh % NH_, kvh = h / (NH_ / NKV_);
  int qr0 = t128 * 128 + wave * 16;

  const unsigned short* Qh = Qb + ((size_t)(b * NH_ + h) * T_) * HD_;
  const unsigned short* Kh = Kb + ((size_t)(b * NKV_ + kvh) * T_) * HD_;
  const unsigned short* Vh = VTb + ((size_t)(b * NKV_ + kvh) * HD_) * T_;

  FragBF aq0, aq1;
  load_a_frag(aq0, Qh, HD_, qr0, 0, lane);
  load_a_frag(aq1, Qh, HD_, qr0, 32, lane);

  v8f o0 = {}, o1 = {}, o2 = {}, o3 = {};
  float mrow[8], lrow[8];
#pragma unroll
  for (int v = 0; v < 8; ++v) { mrow[v] = NEG_INF; lrow[v] = 0.0f; }

  int col = lane & 15, lh = lane >> 4;
  int ktmax = (qr0 + 15) >> 5;          // inclusive; causal
  unsigned short* Pw = &Ptile[wave][0];

  auto load_k_tile = [&](FragBF (&bk)[4], int kbase) {
    load_b_frag(bk[0], Kh, (size_t)HD_, kbase,      0,  lane);
    load_b_frag(bk[1], Kh, (size_t)HD_, kbase,      32, lane);
    load_b_frag(bk[2], Kh, (size_t)HD_, kbase + 16, 0,  lane);
    load_b_frag(bk[3], Kh, (size_t)HD_, kbase + 16, 32, lane);
  };

  FragBF bkc[4], bkn[4];
  load_k_tile(bkc, 0);

  for (int kt = 0; kt <= ktmax; ++kt) {
    int kbase = kt * 32;
    v8f s0 = {}, s1 = {};
    s0 = wmma_bf16(aq0, bkc[0], s0);
    s0 = wmma_bf16(aq1, bkc[1], s0);
    s1 = wmma_bf16(aq0, bkc[2], s1);
    s1 = wmma_bf16(aq1, bkc[3], s1);

    // issue V loads early; softmax VALU below hides their latency
    FragBF bv[4];
#pragma unroll
    for (int j = 0; j < 4; ++j)
      load_b_frag(bv[j], Vh, (size_t)T_, 16 * j, kbase, lane);
    // double-buffer next key tile's K fragments
    if (kt < ktmax) load_k_tile(bkn, kbase + 32);
    // prefetch K/V two tiles ahead (near scope)
    if (kt + 1 < ktmax) {
      __builtin_prefetch(Kh + (size_t)(kbase + 64 + col) * HD_, 0, 3);
      __builtin_prefetch(Vh + (size_t)lane * T_ + kbase + 64, 0, 3);
      __builtin_prefetch(Vh + (size_t)(lane + 32) * T_ + kbase + 64, 0, 3);
    }

    int c0 = kbase + col, c1 = c0 + 16;
#pragma unroll
    for (int v = 0; v < 8; ++v) {
      int rowq = qr0 + v + 8 * lh;
      float sv0 = (c0 <= rowq) ? s0[v] * 0.125f : NEG_INF;
      float sv1 = (c1 <= rowq) ? s1[v] * 0.125f : NEG_INF;
      float smax = fmaxf(sv0, sv1);
#pragma unroll
      for (int off = 8; off >= 1; off >>= 1)
        smax = fmaxf(smax, __shfl_xor(smax, off, 32));
      float mnew = fmaxf(mrow[v], smax);
      float p0 = __expf(sv0 - mnew);
      float p1 = __expf(sv1 - mnew);
      float rs = p0 + p1;
#pragma unroll
      for (int off = 8; off >= 1; off >>= 1) rs += __shfl_xor(rs, off, 32);
      float corr = __expf(mrow[v] - mnew);
      lrow[v] = lrow[v] * corr + rs;
      mrow[v] = mnew;
      o0[v] *= corr; o1[v] *= corr; o2[v] *= corr; o3[v] *= corr;
      int rl = v + 8 * lh;              // C-layout row -> LDS re-stripe for A-frag
      Pw[rl * 32 + col]      = f2bf(p0);
      Pw[rl * 32 + col + 16] = f2bf(p1);
    }
    asm volatile("s_wait_dscnt 0x0" ::: "memory");  // LDS RAW within wave

    FragBF pA;
    load_a_frag(pA, Pw, 32, 0, 0, lane);
    o0 = wmma_bf16(pA, bv[0], o0);
    o1 = wmma_bf16(pA, bv[1], o1);
    o2 = wmma_bf16(pA, bv[2], o2);
    o3 = wmma_bf16(pA, bv[3], o3);

    if (kt < ktmax) {
#pragma unroll
      for (int j = 0; j < 4; ++j) bkc[j] = bkn[j];
    }
  }

#pragma unroll
  for (int v = 0; v < 8; ++v) {
    float inv = 1.0f / lrow[v];
    int rowq = qr0 + v + 8 * lh;
    unsigned short* dst = Y + (size_t)(b * T_ + rowq) * D_ + h * HD_;
    dst[col]      = f2bf(o0[v] * inv);
    dst[col + 16] = f2bf(o1[v] * inv);
    dst[col + 32] = f2bf(o2[v] * inv);
    dst[col + 48] = f2bf(o3[v] * inv);
  }
}

// ---------------- host launch ----------------
extern "C" void kernel_launch(void* const* d_in, const int* in_sizes, int n_in,
                              void* d_out, int out_size, void* d_ws, size_t ws_size,
                              hipStream_t stream) {
  (void)in_sizes; (void)n_in; (void)out_size; (void)ws_size;
  const float* x     = (const float*)d_in[0];
  const float* Wq    = (const float*)d_in[1];
  const float* Wk    = (const float*)d_in[2];
  const float* Wv    = (const float*)d_in[3];
  const float* Wproj = (const float*)d_in[4];
  const float* qg    = (const float*)d_in[5];
  const float* Aq    = (const float*)d_in[6];
  const float* Bq    = (const float*)d_in[7];
  const float* Av    = (const float*)d_in[8];
  const float* Bv    = (const float*)d_in[9];
  float* out = (float*)d_out;

  char* ws = (char*)d_ws;
  size_t off = 0;
  auto carve = [&](size_t bytes) {
    char* p = ws + off;
    off += (bytes + 255) & ~(size_t)255;
    return p;
  };
  unsigned short* xb   = (unsigned short*)carve((size_t)BT_ * D_ * 2);
  unsigned short* Wall = (unsigned short*)carve((size_t)NQKV_ * D_ * 2);
  unsigned short* Wpb  = (unsigned short*)carve((size_t)D_ * D_ * 2);
  float*          qkv  = (float*)carve((size_t)BT_ * NQKV_ * 4);
  unsigned short* Qb   = (unsigned short*)carve((size_t)B_ * NH_ * T_ * HD_ * 2);
  unsigned short* Kb   = (unsigned short*)carve((size_t)B_ * NKV_ * T_ * HD_ * 2);
  unsigned short* VTb  = (unsigned short*)carve((size_t)B_ * NKV_ * HD_ * T_ * 2);
  unsigned short* Yb   = (unsigned short*)carve((size_t)BT_ * D_ * 2);

  {
    size_t n = (size_t)BT_ * D_;
    f32_to_bf16_kernel<<<dim3((unsigned)((n + 255) / 256)), dim3(256), 0, stream>>>(x, xb, n);
  }
  build_wall_kernel<<<dim3((NQKV_ * D_ + 255) / 256), dim3(256), 0, stream>>>(
      Wq, Wk, Wv, Aq, Bq, Av, Bv, Wall);
  {
    size_t n = (size_t)D_ * D_;
    f32_to_bf16_kernel<<<dim3((unsigned)((n + 255) / 256)), dim3(256), 0, stream>>>(Wproj, Wpb, n);
  }
  // fused QKV projection: qkv = x @ Wall^T   [4096 x 1536]
  gemm_bf16_f32<<<dim3(BT_ / 64, NQKV_ / 256), dim3(256), 0, stream>>>(
      xb, Wall, qkv, BT_, NQKV_, D_);
  qkv_postprocess<<<dim3(BT_), dim3(256), 0, stream>>>(qkv, qg, Qb, Kb, VTb);
  flash_attn<<<dim3(B_ * NH_ * (T_ / 128)), dim3(256), 0, stream>>>(Qb, Kb, VTb, Yb);
  // output projection: out = Y @ Wproj^T   [4096 x 1024] fp32
  gemm_bf16_f32<<<dim3(BT_ / 64, D_ / 256), dim3(256), 0, stream>>>(
      Yb, Wpb, out, BT_, D_, D_);
}
